// GCNTriplet_28286654611958
// MI455X (gfx1250) — matrix operand order repaired and checked by reading, not compiled, not run on hardware
//
#include <hip/hip_runtime.h>
#include <math.h>

typedef __attribute__((ext_vector_type(16))) _Float16 v16h;
typedef __attribute__((ext_vector_type(8)))  float    v8f;

#define NNODES 100000
#define NEDGES 3200000
#define FIN    128
#define HDIM   256
#define NGRAPH 64
#define HOUT2  128   // H/2

// ---------------------------------------------------------------- utilities
__global__ void zero_f32_kernel(float* __restrict__ p, long long n) {
  long long i = (long long)blockIdx.x * blockDim.x + threadIdx.x;
  if (i < n) p[i] = 0.0f;
}

// Pack W [K,256] f32 row-major into WMMA B-fragment order (f16):
//   frag = colTile*kTiles + kt ; slot = frag*32 + lane ; elem e
//   value = W[(kt*32 + e + 16*(lane>>4)) * 256 + colTile*16 + (lane&15)]
__global__ void pack_w_kernel(const float* __restrict__ w, _Float16* __restrict__ wpk, int K) {
  int idx = blockIdx.x * blockDim.x + threadIdx.x;     // total = 256*K
  if (idx >= K * HDIM) return;
  int e    = idx & 15;
  int lane = (idx >> 4) & 31;
  int rest = idx >> 9;                                 // colTile*kTiles + kt
  int kTiles = K >> 5;
  int kt = rest % kTiles;
  int c  = rest / kTiles;
  int k   = kt * 32 + e + 16 * (lane >> 4);
  int col = c * 16 + (lane & 15);
  wpk[idx] = (_Float16)w[(size_t)k * HDIM + col];
}

// ------------------------------------------------------------- degree / coef
__global__ void deg_count_kernel(const int* __restrict__ dst, float* __restrict__ deg, int e) {
  int i = blockIdx.x * blockDim.x + threadIdx.x;
  if (i < e) atomicAdd(deg + dst[i], 1.0f);
}

__global__ void make_dinv_kernel(float* __restrict__ d, int n) {
  int i = blockIdx.x * blockDim.x + threadIdx.x;
  if (i < n) d[i] = rsqrtf(d[i] + 1.0f);
}

__global__ void make_coef_kernel(const int* __restrict__ src, const int* __restrict__ dst,
                                 const float* __restrict__ dinv, float* __restrict__ coef, int e) {
  int i = blockIdx.x * blockDim.x + threadIdx.x;
  if (i < e) coef[i] = dinv[src[i]] * dinv[dst[i]];
}

// -------------------------------------------------- WMMA GEMM:  C = A(f32) x Bpk(f16)
// A: [M,K] f32 row-major; Bpk: fragment-packed f16 weights; C: [M,256] f32.
// grid.x = M/16, block = 512 (16 waves; wave w owns the 16x16 tile at cols w*16).
// A tile staged in LDS in fragment order -> each lane reads one aligned v16h.
__global__ void gemm_wmma_f16(const float* __restrict__ A, const _Float16* __restrict__ Bpk,
                              float* __restrict__ C, int K) {
  __shared__ v16h sA[32];                   // 32 lanes x 16 halves, fragment order
  const int lane = threadIdx.x & 31;
  const int wave = threadIdx.x >> 5;        // 0..15 -> column tile
  const int hs   = lane >> 4;
  const int l15  = lane & 15;
  const int rowBase = blockIdx.x * 16;
  const int kTiles  = K >> 5;

  // staging: thread t fills fragment slot (sLane = t>>4, sE = t&15)
  const int sLane = threadIdx.x >> 4;
  const int sE    = threadIdx.x & 15;
  const int sRow  = sLane & 15;
  const int sK    = sE + 16 * (sLane >> 4);
  const float* aBase = A + (size_t)(rowBase + sRow) * K + sK;
  _Float16* sAh = (_Float16*)sA;

  const v16h* Bv = (const v16h*)Bpk;        // fragment slots, 32B each
  const size_t bBase = ((size_t)wave * kTiles) * 32 + lane;

  v8f acc = {};
  for (int kt = 0; kt < kTiles; ++kt) {
    sAh[threadIdx.x] = (_Float16)aBase[kt * 32];
    if (kt + 1 < kTiles) __builtin_prefetch(aBase + (kt + 1) * 32, 0, 1);
    __syncthreads();
    v16h a = sA[lane];                      // 2x ds_load_b128
    v16h b = Bv[bBase + (size_t)kt * 32];   // 2x global_load_b128, coalesced
    acc = __builtin_amdgcn_wmma_f32_16x16x32_f16(false, a, false, b,
                                                 (short)0, acc, false, false);
    __syncthreads();
  }
#pragma unroll
  for (int r = 0; r < 8; ++r)               // D layout: M = r + 8*hs, N = lane&15
    C[(size_t)(rowBase + r + 8 * hs) * HDIM + wave * 16 + l15] = acc[r];
}

// ------------------------------------------------------------- edge scatter
// thread handles 4 consecutive features of one edge: coalesced gather + atomics.
__global__ void scatter_edges_kernel(const float* __restrict__ h, const int* __restrict__ src,
                                     const int* __restrict__ dst, const float* __restrict__ coef,
                                     float* __restrict__ agg) {
  long long idx = (long long)blockIdx.x * blockDim.x + threadIdx.x;
  if (idx >= (long long)NEDGES * (HDIM / 4)) return;
  int e = (int)(idx >> 6);                 // HDIM/4 = 64 chunks per edge
  int f = ((int)(idx & 63)) << 2;
  float c = coef[e];
  const float4 v = *(const float4*)(h + (size_t)src[e] * HDIM + f);
  float* ap = agg + (size_t)dst[e] * HDIM + f;
  atomicAdd(ap + 0, v.x * c);
  atomicAdd(ap + 1, v.y * c);
  atomicAdd(ap + 2, v.z * c);
  atomicAdd(ap + 3, v.w * c);
}

// out = agg + h * dinv^2 + bias  (+ optional relu)
__global__ void finalize_kernel(const float* __restrict__ h, const float* __restrict__ agg,
                                const float* __restrict__ bias, const float* __restrict__ dinv,
                                float* __restrict__ out, int relu) {
  long long idx = (long long)blockIdx.x * blockDim.x + threadIdx.x;
  if (idx >= (long long)NNODES * HDIM) return;
  int row = (int)(idx >> 8);
  int f   = (int)(idx & 255);
  float di = dinv[row];
  float v = agg[idx] + h[idx] * di * di + bias[f];
  out[idx] = relu ? fmaxf(v, 0.0f) : v;
}

// ------------------------------------------------------------------ pooling
__global__ void pool_count_kernel(const int* __restrict__ batch, float* __restrict__ cnt) {
  int i = blockIdx.x * blockDim.x + threadIdx.x;
  if (i < NNODES) atomicAdd(cnt + batch[i], 1.0f);
}

__global__ void pool_sum_kernel(const float* __restrict__ y, const int* __restrict__ batch,
                                float* __restrict__ sums) {
  long long idx = (long long)blockIdx.x * blockDim.x + threadIdx.x;
  if (idx >= (long long)NNODES * HDIM) return;
  int row = (int)(idx >> 8);
  int f   = (int)(idx & 255);
  atomicAdd(sums + (size_t)batch[row] * HDIM + f, y[idx]);
}

// z[g,n] = (sums[g,:] / max(cnt[g],1)) @ lin0_W[:,n] + lin0_b[n]
__global__ void pool_lin_kernel(const float* __restrict__ sums, const float* __restrict__ cnt,
                                const float* __restrict__ w, const float* __restrict__ b,
                                float* __restrict__ z) {
  int idx = blockIdx.x * blockDim.x + threadIdx.x;
  if (idx >= NGRAPH * HOUT2) return;
  int g = idx >> 7, n = idx & 127;
  float inv = 1.0f / fmaxf(cnt[g], 1.0f);
  float acc = 0.0f;
#pragma unroll 4
  for (int k = 0; k < HDIM; ++k) acc += sums[g * HDIM + k] * w[k * HOUT2 + n];
  z[idx] = acc * inv + b[n];
}

// ------------------------------------------------------------------ tail
// d_out layout: z0[8192] z1[8192] z2[8192] correct[1] score_p[64] score_n[64] correct_score[1]
__global__ void tail_kernel(float* __restrict__ out, const float* __restrict__ linW,
                            const float* __restrict__ linb) {
  __shared__ int cc[2];
  int g = threadIdx.x;
  if (g == 0) { cc[0] = 0; cc[1] = 0; }
  __syncthreads();
  const float* z0 = out + (size_t)g * HOUT2;
  const float* z1 = out + 8192 + (size_t)g * HOUT2;
  const float* z2 = out + 16384 + (size_t)g * HOUT2;
  float dp2 = 0.f, dn2 = 0.f, y1 = 0.f, y2 = 0.f;
  const float EPS = 1e-6f;
#pragma unroll 4
  for (int k = 0; k < HOUT2; ++k) {
    float a = z0[k], b = z1[k], c = z2[k];
    float d1 = a - b + EPS; dp2 += d1 * d1;
    float d2 = a - c + EPS; dn2 += d2 * d2;
    float w0 = linW[k], w1 = linW[HOUT2 + k];
    y1 += a * w0 + b * w1;
    y2 += a * w0 + c * w1;
  }
  y1 += linb[0]; y2 += linb[0];
  float sp = 1.0f / (1.0f + expf(-y1));
  float sn = 1.0f / (1.0f + expf(-y2));
  out[24577 + g] = sp;
  out[24641 + g] = sn;
  if (sqrtf(dn2) - sqrtf(dp2) > 0.0f) atomicAdd(&cc[0], 1);
  if (sp - sn > 0.0f)                 atomicAdd(&cc[1], 1);
  __syncthreads();
  if (g == 0) { out[24576] = (float)cc[0]; out[24705] = (float)cc[1]; }
}

// ------------------------------------------------------------------ driver
extern "C" void kernel_launch(void* const* d_in, const int* in_sizes, int n_in,
                              void* d_out, int out_size, void* d_ws, size_t ws_size,
                              hipStream_t stream) {
  (void)in_sizes; (void)n_in; (void)out_size; (void)ws_size;

  const float* xs[3]    = {(const float*)d_in[0], (const float*)d_in[1], (const float*)d_in[2]};
  const int*   eis[3]   = {(const int*)d_in[3], (const int*)d_in[4], (const int*)d_in[5]};
  const int*   batch[3] = {(const int*)d_in[6], (const int*)d_in[7], (const int*)d_in[8]};
  const float* W0 = (const float*)d_in[9];   const float* b0 = (const float*)d_in[10];
  const float* W1 = (const float*)d_in[11];  const float* b1 = (const float*)d_in[12];
  const float* W2 = (const float*)d_in[13];  const float* b2 = (const float*)d_in[14];
  const float* lin0W = (const float*)d_in[15]; const float* lin0b = (const float*)d_in[16];
  const float* linW  = (const float*)d_in[17]; const float* linb  = (const float*)d_in[18];
  float* out = (float*)d_out;

  // workspace layout: packed f16 weights first (keeps 32B alignment for v16h loads)
  _Float16* Wpk0 = (_Float16*)d_ws;                      // [128*256] f16, fragment order
  _Float16* Wpk1 = Wpk0 + FIN * HDIM;                    // [256*256] f16
  _Float16* Wpk2 = Wpk1 + HDIM * HDIM;                   // [256*256] f16
  float* bufA = (float*)(Wpk2 + HDIM * HDIM);            // [N,256]
  float* bufB = bufA + (size_t)NNODES * HDIM;            // [N,256]
  float* dinv = bufB + (size_t)NNODES * HDIM;            // [N]
  float* coef = dinv + NNODES;                           // [E]
  float* sums = coef + NEDGES;                           // [64,256]
  float* cnt  = sums + NGRAPH * HDIM;                    // [64]

  const int B = 256;
  const long long NH = (long long)NNODES * HDIM;
  const int gNH = (int)((NH + B - 1) / B);
  const int gE  = (NEDGES + B - 1) / B;
  const int gN  = (NNODES + B - 1) / B;
  const long long EH4 = (long long)NEDGES * (HDIM / 4);
  const int gEH = (int)((EH4 + B - 1) / B);

  // one-time fragment-order f16 weight packs
  pack_w_kernel<<<(FIN * HDIM + B - 1) / B, B, 0, stream>>>(W0, Wpk0, FIN);
  pack_w_kernel<<<(HDIM * HDIM + B - 1) / B, B, 0, stream>>>(W1, Wpk1, HDIM);
  pack_w_kernel<<<(HDIM * HDIM + B - 1) / B, B, 0, stream>>>(W2, Wpk2, HDIM);

  for (int p = 0; p < 3; ++p) {
    const int* src = eis[p];
    const int* dst = eis[p] + NEDGES;

    // degrees -> dinv -> per-edge coef
    zero_f32_kernel<<<gN, B, 0, stream>>>(dinv, NNODES);
    deg_count_kernel<<<gE, B, 0, stream>>>(dst, dinv, NEDGES);
    make_dinv_kernel<<<gN, B, 0, stream>>>(dinv, NNODES);
    make_coef_kernel<<<gE, B, 0, stream>>>(src, dst, dinv, coef, NEDGES);

    // ---- layer 0: h0 = x @ W0 -> bufA ; agg -> bufB ; y0 -> bufA (relu)
    gemm_wmma_f16<<<NNODES / 16, 512, 0, stream>>>(xs[p], Wpk0, bufA, FIN);
    zero_f32_kernel<<<gNH, B, 0, stream>>>(bufB, NH);
    scatter_edges_kernel<<<gEH, B, 0, stream>>>(bufA, src, dst, coef, bufB);
    finalize_kernel<<<gNH, B, 0, stream>>>(bufA, bufB, b0, dinv, bufA, 1);

    // ---- layer 1: h1 = y0 @ W1 -> bufB ; agg -> bufA ; y1 -> bufB (relu)
    gemm_wmma_f16<<<NNODES / 16, 512, 0, stream>>>(bufA, Wpk1, bufB, HDIM);
    zero_f32_kernel<<<gNH, B, 0, stream>>>(bufA, NH);
    scatter_edges_kernel<<<gEH, B, 0, stream>>>(bufB, src, dst, coef, bufA);
    finalize_kernel<<<gNH, B, 0, stream>>>(bufB, bufA, b1, dinv, bufB, 1);

    // ---- layer 2: h2 = y1 @ W2 -> bufA ; agg -> bufB ; y2 -> bufA (no relu)
    gemm_wmma_f16<<<NNODES / 16, 512, 0, stream>>>(bufB, Wpk2, bufA, HDIM);
    zero_f32_kernel<<<gNH, B, 0, stream>>>(bufB, NH);
    scatter_edges_kernel<<<gEH, B, 0, stream>>>(bufA, src, dst, coef, bufB);
    finalize_kernel<<<gNH, B, 0, stream>>>(bufA, bufB, b2, dinv, bufA, 0);

    // ---- pool + lin0 -> z_p written straight into d_out
    zero_f32_kernel<<<(NGRAPH * HDIM + NGRAPH + B - 1) / B, B, 0, stream>>>(
        sums, NGRAPH * HDIM + NGRAPH);
    pool_count_kernel<<<gN, B, 0, stream>>>(batch[p], cnt);
    pool_sum_kernel<<<gNH, B, 0, stream>>>(bufA, batch[p], sums);
    pool_lin_kernel<<<(NGRAPH * HOUT2 + B - 1) / B, B, 0, stream>>>(
        sums, cnt, lin0W, lin0b, out + (size_t)p * NGRAPH * HOUT2);
  }

  // distances, scores, counts
  tail_kernel<<<1, 64, 0, stream>>>(out, linW, linb);
}
